// MoDGMMFeatureModel_2121713845137
// MI455X (gfx1250) — matrix operand order repaired and checked by reading, not compiled, not run on hardware
//
#include <hip/hip_runtime.h>
#include <math.h>

// ---- types for WMMA ----
typedef _Float16 v16h __attribute__((ext_vector_type(16)));
typedef _Float16 v8h  __attribute__((ext_vector_type(8)));
typedef _Float16 v2h  __attribute__((ext_vector_type(2)));
typedef float    v8f  __attribute__((ext_vector_type(8)));

#define GWD 64
#define GHD 64
#define FEAT 32
#define D_IN 34          // 2 coords + 32 features
#define K1P 40           // strided K for layer-1 staging rows (80B => 16B aligned)
#define D1 128
#define D2 64
#define D3 48
#define WAVES 4
#define BLOCK_T (WAVES*32)
#define PTS_PER_WAVE 16
#define PTS_PER_BLOCK (WAVES*PTS_PER_WAVE)

#define TWO_PI_F      6.2831853071795864769f
#define INV_TWO_PI_F  0.15915494309189533577f

__device__ __forceinline__ v16h vzero16() {
  v16h z;
#pragma unroll
  for (int i = 0; i < 16; i++) z[i] = (_Float16)0.f;
  return z;
}

// A-fragment: per-lane elements j=0..7 are K=(kb*32 + kh*8 + j), j=8..15 are K=+16.
// caller passes base = row + kb*32 + kh*8; the two 8-half chunks are 16B aligned.
__device__ __forceinline__ v16h ldA(const _Float16* base) {
  v8h lo = *(const v8h*)base;
  v8h hi = *(const v8h*)(base + 16);
  return __builtin_shufflevector(lo, hi, 0,1,2,3,4,5,6,7,8,9,10,11,12,13,14,15);
}

// B-fragment: per-lane elements j=0..7 are K=(kb*32 + kh*8 + j), j=8..15 are K=+16.
// caller passes base = Wt_row(n) + kb*32 + kh*8 (16B aligned).
__device__ __forceinline__ v16h ldB(const _Float16* base) {
  v8h lo = *(const v8h*)base;
  v8h hi = *(const v8h*)(base + 16);
  return __builtin_shufflevector(lo, hi, 0,1,2,3,4,5,6,7,8,9,10,11,12,13,14,15);
}

__device__ __forceinline__ v8f splat8(float v) {
  return (v8f){v,v,v,v,v,v,v,v};
}

__device__ __forceinline__ float fast_tanh(float x) {
  // exact in the saturated region; hardware v_exp path
  float xc = fminf(fmaxf(x, -15.f), 15.f);
  float t = __expf(2.f * xc);
  return (t - 1.f) * __builtin_amdgcn_rcpf(t + 1.f);
}

__global__ __launch_bounds__(BLOCK_T)
__attribute__((amdgpu_waves_per_eu(4)))   // cap VGPRs at 256: no s_set_vgpr_msb, >=4 waves/SIMD
void modgmm_kernel(const float* __restrict__ coords, const float* __restrict__ fgrid,
                   const float* __restrict__ W1, const float* __restrict__ B1,
                   const float* __restrict__ G1, const float* __restrict__ Be1,
                   const float* __restrict__ W2, const float* __restrict__ B2,
                   const float* __restrict__ G2, const float* __restrict__ Be2,
                   const float* __restrict__ W3, const float* __restrict__ B3,
                   float* __restrict__ out, int N, int numChunks)
{
  // weights transposed to [n][k] (f16) so B-fragments are contiguous ds_load_b128s
  __shared__ _Float16 sW1t[D1*K1P];          // 10.0 KB
  __shared__ _Float16 sW2t[D2*D1];           // 16.0 KB
  __shared__ _Float16 sW3t[D3*D2];           //  6.0 KB
  __shared__ float sB1[D1], sG1[D1], sBe1[D1];
  __shared__ float sB2[D2], sG2[D2], sBe2[D2];
  __shared__ float sB3[D3];
  __shared__ _Float16 sAct[WAVES][PTS_PER_WAVE*D1]; // 4 KB / wave staging (in, h1, h2)
  __shared__ float    sRaw[WAVES][PTS_PER_WAVE*D3]; // 3 KB / wave raw layer-3 output

  const int tid = threadIdx.x;

  // ---- cooperative weight preload (f32 -> f16, transpose to [n][k]) ----
  for (int i = tid; i < D1*K1P; i += BLOCK_T) {
    int n = i / K1P, k = i - n*K1P;
    sW1t[i] = (_Float16)((k < D_IN) ? W1[k*D1 + n] : 0.f);
  }
  for (int i = tid; i < D2*D1; i += BLOCK_T) {
    int n = i >> 7, k = i & 127;
    sW2t[i] = (_Float16)W2[k*D2 + n];
  }
  for (int i = tid; i < D3*D2; i += BLOCK_T) {
    int n = i >> 6, k = i & 63;
    sW3t[i] = (_Float16)W3[k*D3 + n];
  }
  if (tid < D1) { sB1[tid] = B1[tid]; sG1[tid] = G1[tid]; sBe1[tid] = Be1[tid]; }
  if (tid < D2) { sB2[tid] = B2[tid]; sG2[tid] = G2[tid]; sBe2[tid] = Be2[tid]; }
  if (tid < D3) { sB3[tid] = B3[tid]; }
  __syncthreads();

  const int wv   = tid >> 5;
  const int lane = tid & 31;
  const int m    = lane & 15;   // row in 16-point tile, also N-lane for B/C
  const int kh   = lane >> 4;   // half selector for A/B/C fragment layouts
  const int nl   = m;
  _Float16* act = sAct[wv];
  float*    raw = sRaw[wv];

  for (int chunk = blockIdx.x; chunk < numChunks; chunk += gridDim.x) {
    const int pbase = chunk*PTS_PER_BLOCK + wv*PTS_PER_WAVE;
    const int pg = pbase + m;
    const bool valid = pg < N;
    const int p = valid ? pg : (N - 1);

    { // prefetch next chunk's coords into cache
      int pn = pbase + (int)gridDim.x*PTS_PER_BLOCK + m;
      if (pn < N) __builtin_prefetch(&coords[2*pn], 0, 0);
    }

    // ---- bilinear gather: build f16 input row [cx, cy, feat(32)] in LDS ----
    const float cx = coords[2*p + 0];
    const float cy = coords[2*p + 1];
    float x = cx * (float)(GWD - 1), y = cy * (float)(GHD - 1);
    int x0 = (int)floorf(x); x0 = x0 < 0 ? 0 : (x0 > GWD-2 ? GWD-2 : x0);
    int y0 = (int)floorf(y); y0 = y0 < 0 ? 0 : (y0 > GHD-2 ? GHD-2 : y0);
    const float dx = x - (float)x0, dy = y - (float)y0;
    const float w00 = (1.f-dx)*(1.f-dy), w01 = (1.f-dx)*dy;
    const float w10 = dx*(1.f-dy),       w11 = dx*dy;
    const float4* g00 = (const float4*)(fgrid + (( y0   *GWD + x0  )*FEAT + kh*16));
    const float4* g01 = (const float4*)(fgrid + (((y0+1)*GWD + x0  )*FEAT + kh*16));
    const float4* g10 = (const float4*)(fgrid + (( y0   *GWD + x0+1)*FEAT + kh*16));
    const float4* g11 = (const float4*)(fgrid + (((y0+1)*GWD + x0+1)*FEAT + kh*16));
    _Float16* arow = act + m*K1P;
    if (kh == 0) { v2h c2 = { (_Float16)cx, (_Float16)cy }; *(v2h*)(arow) = c2; }
#pragma unroll
    for (int i = 0; i < 4; i++) {
      float4 a = g00[i], b = g01[i], c = g10[i], d = g11[i];
      float f0 = w00*a.x + w01*b.x + w10*c.x + w11*d.x;
      float f1 = w00*a.y + w01*b.y + w10*c.y + w11*d.y;
      float f2 = w00*a.z + w01*b.z + w10*c.z + w11*d.z;
      float f3 = w00*a.w + w01*b.w + w10*c.w + w11*d.w;
      int e = 2 + kh*16 + i*4;            // even => 4B-aligned v2h stores
      v2h lo = { (_Float16)f0, (_Float16)f1 };
      v2h hi = { (_Float16)f2, (_Float16)f3 };
      *(v2h*)(arow + e)     = lo;
      *(v2h*)(arow + e + 2) = hi;
    }
    // (LDS ops from one wave are in-order: no barrier needed for per-wave staging)

    // ---- layer 1: [16x34] x [34x128], K padded: dense kb=0 + residual K=32,33 ----
    v8f acc1[8];
    {
      v16h afrag = ldA(act + m*K1P + kh*8);
#pragma unroll
      for (int t = 0; t < 8; t++) {
        v16h bfrag = ldB(sW1t + (t*16+nl)*K1P + kh*8);
        acc1[t] = __builtin_amdgcn_wmma_f32_16x16x32_f16(
            false, afrag, false, bfrag, (short)0, splat8(sB1[t*16+nl]), false, false);
      }
    }
    {
      v2h a2 = *(const v2h*)(act + m*K1P + 32);
      v16h afrag = vzero16();
      afrag[0] = kh ? (_Float16)0.f : a2.x;    // K=32
      afrag[1] = kh ? (_Float16)0.f : a2.y;    // K=33
#pragma unroll
      for (int t = 0; t < 8; t++) {
        v2h b2v = *(const v2h*)(sW1t + (t*16+nl)*K1P + 32);
        v16h bfrag = vzero16();
        bfrag[0] = kh ? (_Float16)0.f : b2v.x;
        bfrag[1] = kh ? (_Float16)0.f : b2v.y;
        acc1[t] = __builtin_amdgcn_wmma_f32_16x16x32_f16(
            false, afrag, false, bfrag, (short)0, acc1[t], false, false);
      }
    }

    // ---- relu + layernorm(128), write h1 (f16) back to LDS ----
    {
      float gv[8], bev[8];
#pragma unroll
      for (int t = 0; t < 8; t++) { gv[t] = sG1[t*16+nl]; bev[t] = sBe1[t*16+nl]; }
#pragma unroll
      for (int r = 0; r < 8; r++) {
        float v[8]; float s1 = 0.f, s2 = 0.f;
#pragma unroll
        for (int t = 0; t < 8; t++) {
          float xv = fmaxf(acc1[t][r], 0.f);
          v[t] = xv; s1 += xv; s2 += xv*xv;
        }
#pragma unroll
        for (int msk = 1; msk <= 8; msk <<= 1) {   // reduce within 16-lane half
          s1 += __shfl_xor(s1, msk, 32);
          s2 += __shfl_xor(s2, msk, 32);
        }
        float mean = s1 * (1.f/128.f);
        float var  = fmaxf(s2 * (1.f/128.f) - mean*mean, 0.f);
        float inv  = __builtin_amdgcn_rsqf(var + 1e-5f);
        int row = r + kh*8;
#pragma unroll
        for (int t = 0; t < 8; t++) {
          float yv = (v[t]-mean)*inv*gv[t] + bev[t];
          act[row*D1 + t*16 + nl] = (_Float16)yv;
        }
      }
    }

    // ---- layer 2: [16x128] x [128x64] ----
    v8f acc2[4];
#pragma unroll
    for (int t = 0; t < 4; t++) acc2[t] = splat8(sB2[t*16+nl]);
#pragma unroll
    for (int kb = 0; kb < 4; kb++) {
      v16h afrag = ldA(act + m*D1 + kb*32 + kh*8);
#pragma unroll
      for (int t = 0; t < 4; t++) {
        v16h bfrag = ldB(sW2t + (t*16+nl)*D1 + kb*32 + kh*8);
        acc2[t] = __builtin_amdgcn_wmma_f32_16x16x32_f16(
            false, afrag, false, bfrag, (short)0, acc2[t], false, false);
      }
    }

    // ---- relu + layernorm(64), write h2 (f16) to LDS ----
    {
      float gv[4], bev[4];
#pragma unroll
      for (int t = 0; t < 4; t++) { gv[t] = sG2[t*16+nl]; bev[t] = sBe2[t*16+nl]; }
#pragma unroll
      for (int r = 0; r < 8; r++) {
        float v[4]; float s1 = 0.f, s2 = 0.f;
#pragma unroll
        for (int t = 0; t < 4; t++) {
          float xv = fmaxf(acc2[t][r], 0.f);
          v[t] = xv; s1 += xv; s2 += xv*xv;
        }
#pragma unroll
        for (int msk = 1; msk <= 8; msk <<= 1) {
          s1 += __shfl_xor(s1, msk, 32);
          s2 += __shfl_xor(s2, msk, 32);
        }
        float mean = s1 * (1.f/64.f);
        float var  = fmaxf(s2 * (1.f/64.f) - mean*mean, 0.f);
        float inv  = __builtin_amdgcn_rsqf(var + 1e-5f);
        int row = r + kh*8;
#pragma unroll
        for (int t = 0; t < 4; t++) {
          float yv = (v[t]-mean)*inv*gv[t] + bev[t];
          act[row*D2 + t*16 + nl] = (_Float16)yv;
        }
      }
    }

    // ---- layer 3: [16x64] x [64x48] ----
    v8f acc3[3];
#pragma unroll
    for (int t = 0; t < 3; t++) acc3[t] = splat8(sB3[t*16+nl]);
#pragma unroll
    for (int kb = 0; kb < 2; kb++) {
      v16h afrag = ldA(act + m*D2 + kb*32 + kh*8);
#pragma unroll
      for (int t = 0; t < 3; t++) {
        v16h bfrag = ldB(sW3t + (t*16+nl)*D2 + kb*32 + kh*8);
        acc3[t] = __builtin_amdgcn_wmma_f32_16x16x32_f16(
            false, afrag, false, bfrag, (short)0, acc3[t], false, false);
      }
    }
#pragma unroll
    for (int r = 0; r < 8; r++) {
      int row = r + kh*8;
#pragma unroll
      for (int t = 0; t < 3; t++) raw[row*D3 + t*16 + nl] = acc3[t][r];
    }

    // ---- GMM postprocess: 2 lanes per point (4 components each) ----
    // lane's 24 params are contiguous & 16B aligned: 6x ds_load_b128
    float P[24];
    {
      const float* rh = raw + m*D3 + kh*24;
#pragma unroll
      for (int i = 0; i < 6; i++) {
        float4 q = *(const float4*)(rh + i*4);
        P[i*4+0] = q.x; P[i*4+1] = q.y; P[i*4+2] = q.z; P[i*4+3] = q.w;
      }
    }
#define PP(ci,j) P[(ci)*6+(j)]
    float mx = fmaxf(fmaxf(PP(0,0), PP(1,0)), fmaxf(PP(2,0), PP(3,0)));
    mx = fmaxf(mx, __shfl_xor(mx, 16, 32));
    float ev[4]; float ssum = 0.f;
#pragma unroll
    for (int ci = 0; ci < 4; ci++) { ev[ci] = __expf(PP(ci,0) - mx); ssum += ev[ci]; }
    ssum += __shfl_xor(ssum, 16, 32);
    const float invs = __builtin_amdgcn_rcpf(ssum);

    if (valid) {
#pragma unroll
      for (int ci = 0; ci < 4; ci++) {
        const int c = kh*4 + ci;
        float* o = out + (size_t)p*48 + c*6;
        float ang = PP(ci,2);
        ang = ang - floorf(ang * INV_TWO_PI_F) * TWO_PI_F;     // jnp.mod(x, 2pi)
        float l0 = fminf(fmaxf(PP(ci,3), -10.f), 10.f);
        float l1 = fminf(fmaxf(PP(ci,4), -10.f), 10.f);
        o[0] = ev[ci] * invs;
        o[1] = fmaxf(PP(ci,1), 0.f);
        o[2] = ang;
        o[3] = __expf(l0);
        o[4] = __expf(l1);
        o[5] = 0.99f * fast_tanh(PP(ci,5));
      }
      if (kh == 0) {
        float* oc = out + (size_t)N*48 + (size_t)p*2;
        oc[0] = cx; oc[1] = cy;
      }
    }
#undef PP
  }
}

extern "C" void kernel_launch(void* const* d_in, const int* in_sizes, int n_in,
                              void* d_out, int out_size, void* d_ws, size_t ws_size,
                              hipStream_t stream) {
  (void)n_in; (void)out_size; (void)d_ws; (void)ws_size;
  const float* coords = (const float*)d_in[0];
  const float* fgrid  = (const float*)d_in[1];
  const float* W1  = (const float*)d_in[2];
  const float* B1  = (const float*)d_in[3];
  const float* G1  = (const float*)d_in[4];
  const float* Be1 = (const float*)d_in[5];
  const float* W2  = (const float*)d_in[6];
  const float* B2  = (const float*)d_in[7];
  const float* G2  = (const float*)d_in[8];
  const float* Be2 = (const float*)d_in[9];
  const float* W3  = (const float*)d_in[10];
  const float* B3  = (const float*)d_in[11];
  float* out = (float*)d_out;

  const int N = in_sizes[0] / 2;
  const int numChunks = (N + PTS_PER_BLOCK - 1) / PTS_PER_BLOCK;
  int blocks = numChunks < 2048 ? numChunks : 2048;   // grid-stride amortizes weight preload
  modgmm_kernel<<<blocks, BLOCK_T, 0, stream>>>(coords, fgrid, W1, B1, G1, Be1,
                                                W2, B2, G2, Be2, W3, B3,
                                                out, N, numChunks);
}
